// GATGraphAssociator_89507118449200
// MI455X (gfx1250) — matrix operand order repaired and checked by reading, not compiled, not run on hardware
//
#include <hip/hip_runtime.h>
#include <hip/hip_bf16.h>

typedef __attribute__((ext_vector_type(16))) __bf16 v16bf;
typedef __attribute__((ext_vector_type(8)))  float  v8f;

#define EMBED 128
#define HEADS 16
#define NEG_SLOPE 0.2f
#define GEMM_CHUNK 8           // K-steps staged in LDS per iteration (32 KB)

// ---------------------------------------------------------------- utilities

__global__ void fill_u32_k(unsigned* p, unsigned v, long long n) {
  long long i = (long long)blockIdx.x * blockDim.x + threadIdx.x;
  if (i < n) p[i] = v;
}

__device__ __forceinline__ unsigned f32_key(float f) {
  unsigned b = __float_as_uint(f);
  return (b & 0x80000000u) ? ~b : (b | 0x80000000u);
}
__device__ __forceinline__ float key_f32(unsigned k) {
  unsigned b = (k & 0x80000000u) ? (k & 0x7FFFFFFFu) : ~k;
  return __uint_as_float(b);
}

__device__ __forceinline__ void edge_ends(const int* ei, int E, long long edge,
                                          int& s, int& d) {
  if (edge < E) { s = ei[edge]; d = ei[(long long)E + edge]; }
  else          { s = d = (int)(edge - E); }   // self loop
}

// ------------------------------------------------- WMMA fragment pre-packing
// A fragment (16x32 bf16): lane L -> row M = L%16, group g = L/16
//   elem e<8 : K = g*8 + e ;  elem e>=8 : K = 16 + g*8 + (e-8)
__global__ void pack_a_bf16_k(const float* __restrict__ A, __bf16* __restrict__ out,
                              int M, int K, int MT, int KS) {
  long long t = (long long)blockIdx.x * blockDim.x + threadIdx.x;
  long long total = (long long)MT * KS * 512;
  if (t >= total) return;
  int e    = (int)(t & 15);
  int lane = (int)((t >> 4) & 31);
  long long fi = t >> 9;
  int ks = (int)(fi % KS);
  int mt = (int)(fi / KS);
  int m  = mt * 16 + (lane & 15);
  int g  = lane >> 4;
  int k  = ks * 32 + ((e < 8) ? (g * 8 + e) : (16 + g * 8 + (e - 8)));
  float v = (m < M) ? A[(long long)m * K + k] : 0.0f;
  out[t] = (__bf16)v;
}

// B fragment (32x16 bf16): lane L -> col N = L%16, g = L/16: e<8 : K = g*8 + e ; e>=8 : K = 16 + g*8 + (e-8)
__global__ void pack_b_bf16_k(const float* __restrict__ B, __bf16* __restrict__ out,
                              int K, int Nout, int NT, int KS) {
  long long t = (long long)blockIdx.x * blockDim.x + threadIdx.x;
  long long total = (long long)NT * KS * 512;
  if (t >= total) return;
  int e    = (int)(t & 15);
  int lane = (int)((t >> 4) & 31);
  long long fi = t >> 9;
  int ks = (int)(fi % KS);
  int nt = (int)(fi / KS);
  int n  = nt * 16 + (lane & 15);
  int k  = ks * 32 + ((e < 8) ? ((lane >> 4) * 8 + e) : (16 + (lane >> 4) * 8 + (e - 8)));
  float v = (n < Nout) ? B[(long long)k * Nout + n] : 0.0f;
  out[t] = (__bf16)v;
}

// ------------------------------------------------------------ WMMA GEMM core
// 256-thread block (8 waves): 8 M-tiles x 4 N-tiles macro-tile.
// B fragments for the block's 4 N-tiles are staged in LDS per K-chunk and
// shared by all 8 waves; each wave reuses its A fragment across 4 WMMAs.
__global__ void wmma_gemm_bf16_k(const __bf16* __restrict__ Ap,
                                 const __bf16* __restrict__ Bp,
                                 float* __restrict__ C,
                                 int M, int Nout, int MT, int NT, int KS,
                                 int Nblocks) {
  __shared__ __bf16 Blds[GEMM_CHUNK * 4 * 512];   // 32 KB

  const int wave = threadIdx.x >> 5;
  const int lane = threadIdx.x & 31;
  const int mblk   = blockIdx.x / Nblocks;
  const int ntbase = (blockIdx.x % Nblocks) * 4;
  int mt = mblk * 8 + wave;
  if (mt >= MT) mt = MT - 1;            // clamp: duplicate (identical) work, no divergence

  v8f acc0 = {}, acc1 = {}, acc2 = {}, acc3 = {};

  for (int ks0 = 0; ks0 < KS; ks0 += GEMM_CHUNK) {
    const int cs = (KS - ks0 < GEMM_CHUNK) ? (KS - ks0) : GEMM_CHUNK;

    // cooperative B stage: cs*4 fragments, 64 uint4 each
    const int totalq = cs * 4 * 64;
    for (int q = threadIdx.x; q < totalq; q += blockDim.x) {
      int frag = q >> 6;                // ksc*4 + j
      int r    = q & 63;                // (lane, half)
      int ksc  = frag >> 2;
      int j    = frag & 3;
      int ln   = r >> 1;
      int half = r & 1;
      const uint4* src = (const uint4*)
          (Bp + ((((long long)(ntbase + j)) * KS + ks0 + ksc) * 32 + ln) * 16) + half;
      ((uint4*)Blds)[q] = *src;
    }
    __syncthreads();

    const __bf16* ap = Ap + (((long long)mt * KS + ks0) * 32 + lane) * 16;
    for (int ksc = 0; ksc < cs; ++ksc) {
      v16bf a = *(const v16bf*)ap;
      ap += 512;
      if (ksc + 1 == cs && ks0 + cs < KS) __builtin_prefetch(ap, 0, 0);
      const __bf16* bl = Blds + ((ksc * 4) * 32 + lane) * 16;
      v16bf b0 = *(const v16bf*)(bl);
      v16bf b1 = *(const v16bf*)(bl + 512);
      v16bf b2 = *(const v16bf*)(bl + 1024);
      v16bf b3 = *(const v16bf*)(bl + 1536);
      acc0 = __builtin_amdgcn_wmma_f32_16x16x32_bf16(false, a, false, b0, (short)0, acc0, false, false);
      acc1 = __builtin_amdgcn_wmma_f32_16x16x32_bf16(false, a, false, b1, (short)0, acc1, false, false);
      acc2 = __builtin_amdgcn_wmma_f32_16x16x32_bf16(false, a, false, b2, (short)0, acc2, false, false);
      acc3 = __builtin_amdgcn_wmma_f32_16x16x32_bf16(false, a, false, b3, (short)0, acc3, false, false);
    }
    __syncthreads();
  }

  const int n0    = ntbase * 16 + (lane & 15);
  const int mbase = mt * 16 + ((lane >> 4) << 3);
  #pragma unroll
  for (int r = 0; r < 8; ++r) {
    int m = mbase + r;
    if (m < M) {
      float* cp = C + (long long)m * Nout + n0;
      cp[0]  = acc0[r];
      cp[16] = acc1[r];
      cp[32] = acc2[r];
      cp[48] = acc3[r];
    }
  }
}

// ---------------------------------------------------------------- GAT pieces

// per-(node,head) attention logits: als = h.a_src, ald = h.a_dst
__global__ void attn_logits_k(const float* __restrict__ Hm,
                              const float* __restrict__ a_src,
                              const float* __restrict__ a_dst,
                              float* __restrict__ als, float* __restrict__ ald,
                              int N, int Hh) {
  long long t = (long long)blockIdx.x * blockDim.x + threadIdx.x;
  if (t >= (long long)N * Hh) return;
  int h = (int)(t % Hh);
  const float* hp = Hm + t * EMBED;
  const float* as = a_src + (long long)h * EMBED;
  const float* ad = a_dst + (long long)h * EMBED;
  float s = 0.0f, d = 0.0f;
  #pragma unroll 4
  for (int c = 0; c < EMBED; ++c) { float v = hp[c]; s += v * as[c]; d += v * ad[c]; }
  als[t] = s; ald[t] = d;
}

// phase 1: e = leaky_relu(als[src]+ald[dst]); segment max into mkey[dst]
__global__ void edge_logit_max_k(const int* __restrict__ ei, int E, int Hh,
                                 const float* __restrict__ als,
                                 const float* __restrict__ ald,
                                 float* __restrict__ ebuf,
                                 unsigned* __restrict__ mkey, long long EtotH) {
  long long t = (long long)blockIdx.x * blockDim.x + threadIdx.x;
  if (t >= EtotH) return;
  long long edge = t / Hh;
  int h = (int)(t % Hh);
  int s, d; edge_ends(ei, E, edge, s, d);
  float v = als[(long long)s * Hh + h] + ald[(long long)d * Hh + h];
  v = (v > 0.0f) ? v : NEG_SLOPE * v;
  ebuf[t] = v;
  atomicMax(&mkey[(long long)d * Hh + h], f32_key(v));
}

// phase 2: ex = exp(e - m[dst]); segment sum into z[dst]
__global__ void edge_exp_sum_k(const int* __restrict__ ei, int E, int Hh,
                               float* __restrict__ ebuf,
                               const unsigned* __restrict__ mkey,
                               float* __restrict__ z, long long EtotH) {
  long long t = (long long)blockIdx.x * blockDim.x + threadIdx.x;
  if (t >= EtotH) return;
  long long edge = t / Hh;
  int h = (int)(t % Hh);
  int s, d; edge_ends(ei, E, edge, s, d);
  float m  = key_f32(mkey[(long long)d * Hh + h]);
  float ex = __expf(ebuf[t] - m);
  ebuf[t] = ex;
  atomicAdd(&z[(long long)d * Hh + h], ex);
}

// phase 3: agg[dst,h,:] += h[src,h,:] * alpha
// thread = one (edge, head, float4): b128 gather + 4 L2 atomics
__global__ void aggregate_k(const int* __restrict__ ei, int E, int Hh,
                            const float* __restrict__ ebuf,
                            const float* __restrict__ z,
                            const float* __restrict__ Hm,
                            float* __restrict__ agg, long long total) {
  long long t = (long long)blockIdx.x * blockDim.x + threadIdx.x;
  if (t >= total) return;
  int c4 = (int)(t & 31);               // EMBED/4 = 32 quads
  long long eh = t >> 5;
  long long edge = eh / Hh;
  int h = (int)(eh % Hh);
  int s, d; edge_ends(ei, E, edge, s, d);
  float alpha = ebuf[eh] / (z[(long long)d * Hh + h] + 1e-16f);
  const float4 v = *(const float4*)(Hm + ((long long)s * Hh + h) * EMBED + c4 * 4);
  float* ap = agg + ((long long)d * Hh + h) * EMBED + c4 * 4;
  atomicAdd(ap + 0, v.x * alpha);
  atomicAdd(ap + 1, v.y * alpha);
  atomicAdd(ap + 2, v.z * alpha);
  atomicAdd(ap + 3, v.w * alpha);
}

// agg = elu(agg + b)
__global__ void elu_bias_k(float* __restrict__ agg, const float* __restrict__ b,
                           int cols, long long total) {
  long long i = (long long)blockIdx.x * blockDim.x + threadIdx.x;
  if (i >= total) return;
  float v = agg[i] + b[i % cols];
  agg[i] = (v > 0.0f) ? v : (__expf(v) - 1.0f);
}

// out[n] = (agg2[n,:] + b2) . Wc + bc
__global__ void classifier_k(const float* __restrict__ agg2,
                             const float* __restrict__ b2,
                             const float* __restrict__ Wc,
                             const float* __restrict__ bc,
                             float* __restrict__ out, int N) {
  int n = blockIdx.x * blockDim.x + threadIdx.x;
  if (n >= N) return;
  const float* hp = agg2 + (long long)n * EMBED;
  float acc = 0.0f;
  #pragma unroll 4
  for (int c = 0; c < EMBED; ++c) acc += (hp[c] + b2[c]) * Wc[c];
  out[n] = acc + bc[0];
}

// ------------------------------------------------------------------- driver

static inline unsigned gridFor(long long n, int block) {
  return (unsigned)((n + block - 1) / block);
}

extern "C" void kernel_launch(void* const* d_in, const int* in_sizes, int n_in,
                              void* d_out, int out_size, void* d_ws, size_t ws_size,
                              hipStream_t stream) {
  const float* x      = (const float*)d_in[0];
  const int*   ei     = (const int*)  d_in[1];
  const float* W1     = (const float*)d_in[2];
  const float* a_src1 = (const float*)d_in[3];
  const float* a_dst1 = (const float*)d_in[4];
  const float* b1     = (const float*)d_in[5];
  const float* W2     = (const float*)d_in[6];
  const float* a_src2 = (const float*)d_in[7];
  const float* a_dst2 = (const float*)d_in[8];
  const float* b2     = (const float*)d_in[9];
  const float* Wc     = (const float*)d_in[10];
  const float* bc     = (const float*)d_in[11];
  float* out = (float*)d_out;

  const int N    = in_sizes[0] / EMBED;     // 10000
  const int E    = in_sizes[1] / 2;         // 160000
  const int Etot = E + N;                   // + self loops

  const int MT  = (N + 15) / 16;            // 625
  const int K1  = EMBED,         KS1 = K1 / 32, Nout1 = HEADS * EMBED, NT1 = Nout1 / 16;
  const int K2  = HEADS * EMBED, KS2 = K2 / 32, Nout2 = EMBED,         NT2 = Nout2 / 16;
  const int Mblk  = (MT + 7) / 8;
  const int Nblk1 = NT1 / 4;                // 32
  const int Nblk2 = NT2 / 4;                // 2

  // ---- bump allocator over d_ws
  char* wsp = (char*)d_ws;
  size_t off = 0;
  auto alloc = [&](size_t bytes) -> char* {
    char* p = wsp + off;
    off += (bytes + 255) & ~(size_t)255;
    return p;
  };

  __bf16*   Apack1 = (__bf16*)  alloc((size_t)MT  * KS1 * 512 * 2);
  __bf16*   Bpack1 = (__bf16*)  alloc((size_t)NT1 * KS1 * 512 * 2);
  float*    H1     = (float*)   alloc((size_t)N * Nout1 * 4);     // reused for Apack2
  float*    als1   = (float*)   alloc((size_t)N * HEADS * 4);
  float*    ald1   = (float*)   alloc((size_t)N * HEADS * 4);
  unsigned* m1     = (unsigned*)alloc((size_t)N * HEADS * 4);
  float*    z1     = (float*)   alloc((size_t)N * HEADS * 4);
  float*    e1     = (float*)   alloc((size_t)Etot * HEADS * 4);  // reused as e2
  float*    agg1   = (float*)   alloc((size_t)N * Nout1 * 4);
  __bf16*   Bpack2 = (__bf16*)  alloc((size_t)NT2 * KS2 * 512 * 2);
  float*    H2     = (float*)   alloc((size_t)N * EMBED * 4);
  float*    als2   = (float*)   alloc((size_t)N * 4);
  float*    ald2   = (float*)   alloc((size_t)N * 4);
  unsigned* m2     = (unsigned*)alloc((size_t)N * 4);
  float*    z2     = (float*)   alloc((size_t)N * 4);
  float*    agg2   = (float*)   alloc((size_t)N * EMBED * 4);
  __bf16*   Apack2 = (__bf16*)H1;                                 // alias (H1 dead by then)
  float*    e2     = e1;                                          // alias
  (void)ws_size; (void)n_in; (void)out_size;

  const int B = 256;
  const unsigned NEG_INF_KEY = 0x007FFFFFu;   // f32_key(-inf)

  // ---- init (deterministic per call)
  fill_u32_k<<<gridFor((long long)N * Nout1, B), B, 0, stream>>>((unsigned*)agg1, 0u, (long long)N * Nout1);
  fill_u32_k<<<gridFor((long long)N * HEADS, B), B, 0, stream>>>((unsigned*)z1, 0u, (long long)N * HEADS);
  fill_u32_k<<<gridFor((long long)N * HEADS, B), B, 0, stream>>>(m1, NEG_INF_KEY, (long long)N * HEADS);
  fill_u32_k<<<gridFor((long long)N * EMBED, B), B, 0, stream>>>((unsigned*)agg2, 0u, (long long)N * EMBED);
  fill_u32_k<<<gridFor(N, B), B, 0, stream>>>((unsigned*)z2, 0u, N);
  fill_u32_k<<<gridFor(N, B), B, 0, stream>>>(m2, NEG_INF_KEY, N);

  // ---- layer 1: H1 = x @ W1 (bf16 WMMA, LDS-staged B)
  pack_a_bf16_k<<<gridFor((long long)MT * KS1 * 512, B), B, 0, stream>>>(x, Apack1, N, K1, MT, KS1);
  pack_b_bf16_k<<<gridFor((long long)NT1 * KS1 * 512, B), B, 0, stream>>>(W1, Bpack1, K1, Nout1, NT1, KS1);
  wmma_gemm_bf16_k<<<(unsigned)(Mblk * Nblk1), B, 0, stream>>>(Apack1, Bpack1, H1, N, Nout1, MT, NT1, KS1, Nblk1);

  attn_logits_k<<<gridFor((long long)N * HEADS, B), B, 0, stream>>>(H1, a_src1, a_dst1, als1, ald1, N, HEADS);

  long long EH1 = (long long)Etot * HEADS;
  edge_logit_max_k<<<gridFor(EH1, B), B, 0, stream>>>(ei, E, HEADS, als1, ald1, e1, m1, EH1);
  edge_exp_sum_k<<<gridFor(EH1, B), B, 0, stream>>>(ei, E, HEADS, e1, m1, z1, EH1);
  aggregate_k<<<gridFor(EH1 * 32, B), B, 0, stream>>>(ei, E, HEADS, e1, z1, H1, agg1, EH1 * 32);

  elu_bias_k<<<gridFor((long long)N * Nout1, B), B, 0, stream>>>(agg1, b1, Nout1, (long long)N * Nout1);

  // ---- layer 2: H2 = agg1 @ W2 (bf16 WMMA); Apack2 aliases dead H1
  pack_a_bf16_k<<<gridFor((long long)MT * KS2 * 512, B), B, 0, stream>>>(agg1, Apack2, N, K2, MT, KS2);
  pack_b_bf16_k<<<gridFor((long long)NT2 * KS2 * 512, B), B, 0, stream>>>(W2, Bpack2, K2, Nout2, NT2, KS2);
  wmma_gemm_bf16_k<<<(unsigned)(Mblk * Nblk2), B, 0, stream>>>(Apack2, Bpack2, H2, N, Nout2, MT, NT2, KS2, Nblk2);

  attn_logits_k<<<gridFor(N, B), B, 0, stream>>>(H2, a_src2, a_dst2, als2, ald2, N, 1);

  long long EH2 = (long long)Etot;
  edge_logit_max_k<<<gridFor(EH2, B), B, 0, stream>>>(ei, E, 1, als2, ald2, e2, m2, EH2);
  edge_exp_sum_k<<<gridFor(EH2, B), B, 0, stream>>>(ei, E, 1, e2, m2, z2, EH2);
  aggregate_k<<<gridFor(EH2 * 32, B), B, 0, stream>>>(ei, E, 1, e2, z2, H2, agg2, EH2 * 32);

  // ---- classifier
  classifier_k<<<gridFor(N, B), B, 0, stream>>>(agg2, b2, Wc, bc, out, N);
}
